// DyIntraModalityUpdate_343597383850
// MI455X (gfx1250) — matrix operand order repaired and checked by reading, not compiled, not run on hardware
//
#include <hip/hip_runtime.h>
#include <hip/hip_bf16.h>

// ---------------- constants ----------------
constexpr int kB  = 16;     // batch
constexpr int kS  = 512;    // seq len (NO = LQ = LC)
constexpr int kD  = 1024;   // model dim
constexpr int kH  = 8;      // heads
constexpr int kDH = 128;    // head dim
constexpr int kTD = 3 * kD; // 3072
constexpr int kBS = kB * kS;       // 8192 rows
constexpr int kSKP_VC = 544;       // 513 padded to multiple of 32
constexpr int kSKP_Q  = 512;

typedef __bf16 bf16;
typedef __attribute__((ext_vector_type(16))) __bf16 v16bf;
typedef __attribute__((ext_vector_type(8)))  float  v8f;

// ---------------- small prep kernels ----------------

// masked mean over sequence: mean[b,d] = sum_s x[b,s,d]*m[b,s] / sum_s m[b,s]
__global__ void k_mean(const float* __restrict__ x, const float* __restrict__ mask,
                       float* __restrict__ mean) {
  int idx = blockIdx.x * blockDim.x + threadIdx.x;
  if (idx >= kB * kD) return;
  int b = idx / kD, d = idx % kD;
  float s = 0.f, ms = 0.f;
  for (int t = 0; t < kS; ++t) {
    float m = mask[b * kS + t];
    s += x[((size_t)b * kS + t) * kD + d] * m;
    ms += m;
  }
  mean[idx] = s / ms;
}

// gate[b,j] = sigmoid( relu(mean[b,:]) @ W[:,j] + bias[j] )
__global__ void k_gate(const float* __restrict__ mean, const float* __restrict__ W,
                       const float* __restrict__ bias, float* __restrict__ gate) {
  int idx = blockIdx.x * blockDim.x + threadIdx.x;
  if (idx >= kB * kD) return;
  int b = idx / kD, j = idx % kD;
  float acc = bias[j];
  for (int k = 0; k < kD; ++k)
    acc += fmaxf(mean[b * kD + k], 0.f) * W[(size_t)k * kD + j];
  gate[idx] = 1.f / (1.f + __expf(-acc));
}

// dst[i] = bf16(relu(src[i]))
__global__ void k_relu_bf16(const float* __restrict__ src, bf16* __restrict__ dst, int n) {
  int i = blockIdx.x * blockDim.x + threadIdx.x;
  if (i < n) dst[i] = (bf16)fmaxf(src[i], 0.f);
}

// Wt[j,k] = bf16(W[k,j]);  W is [K,N] row-major, Wt is [N,K] row-major
__global__ void k_transpose_bf16(const float* __restrict__ W, bf16* __restrict__ Wt,
                                 int K, int N) {
  int idx = blockIdx.x * blockDim.x + threadIdx.x;
  if (idx >= K * N) return;
  int k = idx / N, j = idx % N;
  Wt[(size_t)j * K + k] = (bf16)W[idx];
}

// memory token (row 0 of K / col 0 of V^T) + zero the pad rows
__global__ void k_memtok(bf16* __restrict__ Kst, bf16* __restrict__ Vt,
                         const float* __restrict__ mk, const float* __restrict__ mv,
                         const float* __restrict__ gate, int SKP) {
  int idx = blockIdx.x * blockDim.x + threadIdx.x;
  if (idx >= kB * kD) return;
  int b = idx / kD, cch = idx % kD;
  int h = cch >> 7, d = cch & 127;
  float g = gate ? (1.f + gate[b * kD + cch]) : 1.f;
  size_t bh = (size_t)(b * kH + h);
  Kst[(bh * SKP + 0) * kDH + d] = (bf16)(32.f * mk[cch] * g);     // sqrt(D)=32
  Vt[(bh * kDH + d) * (size_t)SKP + 0] = (bf16)mv[cch];
  for (int s = kS + 1; s < SKP; ++s) {
    Kst[(bh * SKP + s) * kDH + d] = (bf16)0.f;
    Vt[(bh * kDH + d) * (size_t)SKP + s] = (bf16)0.f;
  }
}

// key mask with optional memory token at 0 and zero padding
__global__ void k_kmask(float* __restrict__ km, const float* __restrict__ mask,
                        int SKP, int hasMem) {
  int idx = blockIdx.x * blockDim.x + threadIdx.x;
  if (idx >= kB * SKP) return;
  int b = idx / SKP, i = idx % SKP;
  float m;
  if (hasMem) m = (i == 0) ? 1.f : ((i - 1) < kS ? mask[b * kS + i - 1] : 0.f);
  else        m = (i < kS) ? mask[b * kS + i] : 0.f;
  km[idx] = m;
}

// ---------------- WMMA fragment helpers ----------------
// A fragment: 16x32 bf16, row m=lane&15, element e -> K = (e>>3)*16 + (lane>>4)*8 + (e&7)
// -> two contiguous 16B loads at row + hh*8 and +16.
__device__ __forceinline__ v16bf load_frag_rowK(const bf16* __restrict__ p16) {
  v16bf f;
  ((uint4*)&f)[0] = *(const uint4*)(p16);
  ((uint4*)&f)[1] = *(const uint4*)(p16 + 16);
  return f;
}
// B fragment: 32x16 bf16, col n=lane&15, element e -> K = (e>>3)*16 + (lane>>4)*8 + (e&7)
// -> two contiguous 16B loads at col + hh*8 and +16.
__device__ __forceinline__ v16bf load_frag_colK(const bf16* __restrict__ p16) {
  v16bf f;
  ((uint4*)&f)[0] = *(const uint4*)(p16);
  ((uint4*)&f)[1] = *(const uint4*)(p16 + 16);
  return f;
}

// ---------------- fused QKV projection GEMM (WMMA) ----------------
// C[8192,3072] = Abf[8192,1024] @ W ;  Wt is W transposed [3072,1024] bf16.
// epilogue: +bias, *seqmask, *(1+gate) on k/q thirds, scatter into per-head K/Q/V^T.
__global__ __launch_bounds__(256)
void k_gemm_qkv(const bf16* __restrict__ A, const bf16* __restrict__ Wt,
                const float* __restrict__ bias, const float* __restrict__ mask,
                const float* __restrict__ gate,   // nullptr => ungated (c modality)
                bf16* __restrict__ Kst, bf16* __restrict__ Qst, bf16* __restrict__ Vt,
                int SKP, int koff) {
  const int lane = threadIdx.x & 31;
  const int wave = threadIdx.x >> 5;            // 0..7
  const int wm = wave >> 2, wn = wave & 3;      // 2(M) x 4(N) waves
  const int m0 = blockIdx.y * 32 + wm * 16;
  const int n0 = blockIdx.x * 256 + wn * 64;
  const int nl = lane & 15, hh = lane >> 4;

  v8f acc[4];
  #pragma unroll
  for (int nt = 0; nt < 4; ++nt)
    #pragma unroll
    for (int r = 0; r < 8; ++r) acc[nt][r] = 0.f;

  const bf16* Arow = A + (size_t)(m0 + nl) * kD + hh * 8;
  for (int kc = 0; kc < kD; kc += 32) {
    v16bf af = load_frag_rowK(Arow + kc);
    #pragma unroll
    for (int nt = 0; nt < 4; ++nt) {
      const bf16* bp = Wt + (size_t)(n0 + nt * 16 + nl) * kD + kc + hh * 8;
      v16bf bfg = load_frag_colK(bp);
      acc[nt] = __builtin_amdgcn_wmma_f32_16x16x32_bf16(
          false, af, false, bfg, (short)0, acc[nt], false, false);
    }
  }

  #pragma unroll
  for (int nt = 0; nt < 4; ++nt) {
    int j = n0 + nt * 16 + nl;          // global output column [0,3072)
    int part = j >> 10;                 // 0=k 1=q 2=v  (split order k,q,v)
    int cch = j & 1023;
    int h = cch >> 7, d = cch & 127;
    float bv = bias[j];
    #pragma unroll
    for (int r = 0; r < 8; ++r) {
      int mg = m0 + r + 8 * hh;         // global row in [0,8192)
      int b = mg >> 9, s = mg & 511;
      float val = acc[nt][r] + bv;
      val *= mask[b * kS + s];
      if (gate && part < 2) val *= (1.f + gate[b * kD + cch]);
      size_t bh = (size_t)(b * kH + h);
      if (part == 0)      Kst[(bh * SKP + s + koff) * kDH + d] = (bf16)val;
      else if (part == 1) Qst[(bh * (size_t)kS + s) * kDH + d] = (bf16)val;
      else                Vt [(bh * kDH + d) * (size_t)SKP + s + koff] = (bf16)val;
    }
  }
}

// ---------------- flash attention (WMMA) ----------------
// grid: (S/64, H, B), 128 threads (4 waves); wave w owns q rows [q0, q0+16)
__global__ __launch_bounds__(128)
void k_attn(const bf16* __restrict__ Qst, const bf16* __restrict__ Kst,
            const bf16* __restrict__ Vt, const float* __restrict__ km,
            const float* __restrict__ xin, bf16* __restrict__ resid, int SKP) {
  __shared__ __align__(16) float s_sc[4][16][32];
  __shared__ __align__(16) bf16  s_p[4][16][32];
  __shared__ float s_scale[4][16];
  __shared__ float s_l[4][16];

  const int lane = threadIdx.x & 31, w = threadIdx.x >> 5;
  const int b = blockIdx.z, h = blockIdx.y;
  const int q0 = blockIdx.x * 64 + w * 16;
  const int nl = lane & 15, hh = lane >> 4;
  const float rsd = 0.08838834764831845f; // 1/sqrt(128)

  const size_t bh = (size_t)(b * kH + h);
  const bf16* Qb = Qst + (bh * kS + q0) * kDH;
  const bf16* Kb = Kst + bh * (size_t)SKP * kDH;
  const bf16* Vb = Vt + bh * (size_t)kDH * SKP;
  const float* kmb = km + b * SKP;

  v16bf qf[4];
  #pragma unroll
  for (int ch = 0; ch < 4; ++ch)
    qf[ch] = load_frag_rowK(Qb + (size_t)nl * kDH + ch * 32 + hh * 8);

  v8f out[8];
  #pragma unroll
  for (int nt = 0; nt < 8; ++nt)
    #pragma unroll
    for (int r = 0; r < 8; ++r) out[nt][r] = 0.f;
  float m_i = -1e30f, l_i = 0.f;

  for (int kt = 0; kt < SKP; kt += 32) {
    // scores: two 16x16 tiles (keys kt..kt+15, kt+16..kt+31)
    #pragma unroll
    for (int t = 0; t < 2; ++t) {
      v8f sc;
      #pragma unroll
      for (int r = 0; r < 8; ++r) sc[r] = 0.f;
      #pragma unroll
      for (int ch = 0; ch < 4; ++ch) {
        // B = K^T chunk: b[e] = K[kt+t*16+n][ch*32 + (e>>3)*16 + hh*8 + (e&7)]
        const bf16* bp = Kb + (size_t)(kt + t * 16 + nl) * kDH + ch * 32 + hh * 8;
        v16bf bfg = load_frag_colK(bp);
        sc = __builtin_amdgcn_wmma_f32_16x16x32_bf16(
            false, qf[ch], false, bfg, (short)0, sc, false, false);
      }
      #pragma unroll
      for (int r = 0; r < 8; ++r) s_sc[w][r + 8 * hh][t * 16 + nl] = sc[r];
    }

    // online softmax: lanes 0..15 each own one q row
    if (lane < 16) {
      float mx = m_i;
      #pragma unroll
      for (int cc = 0; cc < 32; ++cc) {
        float v = s_sc[w][lane][cc];
        float kv = kmb[kt + cc];
        v = (kv == 0.f ? -1e9f : v) * rsd;   // mask first, then /sqrt(dh) (as in ref)
        s_sc[w][lane][cc] = v;
        mx = fmaxf(mx, v);
      }
      float scale = __expf(m_i - mx);
      float sum = 0.f;
      #pragma unroll
      for (int cc = 0; cc < 32; ++cc) {
        float e = __expf(s_sc[w][lane][cc] - mx);
        sum += e;
        s_p[w][lane][cc] = (bf16)e;
      }
      l_i = l_i * scale + sum;
      m_i = mx;
      s_scale[w][lane] = scale;
    }

    // rescale accumulators by the per-row correction factor
    #pragma unroll
    for (int r = 0; r < 8; ++r) {
      float sc2 = s_scale[w][r + 8 * hh];
      #pragma unroll
      for (int nt = 0; nt < 8; ++nt) out[nt][r] *= sc2;
    }

    // P (16x32 bf16) as A fragment from LDS
    v16bf pf = load_frag_rowK(&s_p[w][nl][hh * 8]);

    // out += P @ V : b[e] = V[kt + (e>>3)*16 + hh*8 + (e&7)][nt*16+n] = Vt[(nt*16+n)*SKP + ...]
    #pragma unroll
    for (int nt = 0; nt < 8; ++nt) {
      const bf16* bp = Vb + (size_t)(nt * 16 + nl) * SKP + kt + hh * 8;
      v16bf bfg = load_frag_colK(bp);
      out[nt] = __builtin_amdgcn_wmma_f32_16x16x32_bf16(
          false, pf, false, bfg, (short)0, out[nt], false, false);
    }
  }

  if (lane < 16) s_l[w][lane] = l_i;

  // normalize, fuse residual, emit bf16 for the output projection
  #pragma unroll
  for (int r = 0; r < 8; ++r) {
    int qrow = q0 + r + 8 * hh;
    float linv = 1.f / s_l[w][r + 8 * hh];
    size_t baseo = ((size_t)b * kS + qrow) * kD + h * kDH;
    #pragma unroll
    for (int nt = 0; nt < 8; ++nt) {
      int d = nt * 16 + nl;
      resid[baseo + d] = (bf16)(xin[baseo + d] + out[nt][r] * linv);
    }
  }
}

// ---------------- output projection GEMM (WMMA) ----------------
// Y[8192,1024] = residbf @ Wout + bias  (f32 into d_out section)
__global__ __launch_bounds__(256)
void k_gemm_out(const bf16* __restrict__ A, const bf16* __restrict__ Wt,
                const float* __restrict__ bias, float* __restrict__ Y) {
  const int lane = threadIdx.x & 31;
  const int wave = threadIdx.x >> 5;
  const int wm = wave >> 2, wn = wave & 3;
  const int m0 = blockIdx.y * 32 + wm * 16;
  const int n0 = blockIdx.x * 256 + wn * 64;
  const int nl = lane & 15, hh = lane >> 4;

  v8f acc[4];
  #pragma unroll
  for (int nt = 0; nt < 4; ++nt)
    #pragma unroll
    for (int r = 0; r < 8; ++r) acc[nt][r] = 0.f;

  const bf16* Arow = A + (size_t)(m0 + nl) * kD + hh * 8;
  for (int kc = 0; kc < kD; kc += 32) {
    v16bf af = load_frag_rowK(Arow + kc);
    #pragma unroll
    for (int nt = 0; nt < 4; ++nt) {
      const bf16* bp = Wt + (size_t)(n0 + nt * 16 + nl) * kD + kc + hh * 8;
      v16bf bfg = load_frag_colK(bp);
      acc[nt] = __builtin_amdgcn_wmma_f32_16x16x32_bf16(
          false, af, false, bfg, (short)0, acc[nt], false, false);
    }
  }
  #pragma unroll
  for (int nt = 0; nt < 4; ++nt) {
    int j = n0 + nt * 16 + nl;
    float bv = bias[j];
    #pragma unroll
    for (int r = 0; r < 8; ++r) {
      int mg = m0 + r + 8 * hh;
      Y[(size_t)mg * kD + j] = acc[nt][r] + bv;
    }
  }
}

// ---------------- host orchestration ----------------
extern "C" void kernel_launch(void* const* d_in, const int* in_sizes, int n_in,
                              void* d_out, int out_size, void* d_ws, size_t ws_size,
                              hipStream_t stream) {
  const float* v       = (const float*)d_in[0];
  const float* q       = (const float*)d_in[1];
  const float* c       = (const float*)d_in[2];
  const float* v_mask  = (const float*)d_in[3];
  const float* q_mask  = (const float*)d_in[4];
  const float* c_mask  = (const float*)d_in[5];
  const float* v4q_w   = (const float*)d_in[6];
  const float* v4q_b   = (const float*)d_in[7];
  const float* q4v_w   = (const float*)d_in[8];
  const float* q4v_b   = (const float*)d_in[9];
  const float* v_lin_w = (const float*)d_in[10];
  const float* v_lin_b = (const float*)d_in[11];
  const float* q_lin_w = (const float*)d_in[12];
  const float* q_lin_b = (const float*)d_in[13];
  const float* c_lin_w = (const float*)d_in[14];
  const float* c_lin_b = (const float*)d_in[15];
  const float* m_v_k   = (const float*)d_in[16];
  const float* m_v_v   = (const float*)d_in[17];
  const float* m_c_k   = (const float*)d_in[18];
  const float* m_c_v   = (const float*)d_in[19];
  const float* v_out_w = (const float*)d_in[20];
  const float* v_out_b = (const float*)d_in[21];
  const float* q_out_w = (const float*)d_in[22];
  const float* q_out_b = (const float*)d_in[23];
  const float* c_out_w = (const float*)d_in[24];
  const float* c_out_b = (const float*)d_in[25];
  (void)in_sizes; (void)n_in; (void)out_size; (void)ws_size;

  char* base = (char*)d_ws;
  size_t off = 0;
  auto take = [&](size_t bytes) -> char* {
    char* p = base + off;
    off = (off + bytes + 255) & ~(size_t)255;
    return p;
  };

  float* vmean = (float*)take(kB * kD * 4);
  float* qmean = (float*)take(kB * kD * 4);
  float* g_v4q = (float*)take(kB * kD * 4);   // gates q's q/k
  float* g_q4v = (float*)take(kB * kD * 4);   // gates v's q/k

  bf16* act_v = (bf16*)take((size_t)kBS * kD * 2);
  bf16* act_q = (bf16*)take((size_t)kBS * kD * 2);
  bf16* act_c = (bf16*)take((size_t)kBS * kD * 2);

  bf16* Wt_v  = (bf16*)take((size_t)kTD * kD * 2);
  bf16* Wt_q  = (bf16*)take((size_t)kTD * kD * 2);
  bf16* Wt_c  = (bf16*)take((size_t)kTD * kD * 2);
  bf16* Wt_vo = (bf16*)take((size_t)kD * kD * 2);
  bf16* Wt_qo = (bf16*)take((size_t)kD * kD * 2);
  bf16* Wt_co = (bf16*)take((size_t)kD * kD * 2);

  bf16* K_v = (bf16*)take((size_t)kB * kH * kSKP_VC * kDH * 2);
  bf16* Q_v = (bf16*)take((size_t)kB * kH * kS * kDH * 2);
  bf16* V_v = (bf16*)take((size_t)kB * kH * kSKP_VC * kDH * 2);
  bf16* K_q = (bf16*)take((size_t)kB * kH * kSKP_Q * kDH * 2);
  bf16* Q_q = (bf16*)take((size_t)kB * kH * kS * kDH * 2);
  bf16* V_q = (bf16*)take((size_t)kB * kH * kSKP_Q * kDH * 2);
  bf16* K_c = (bf16*)take((size_t)kB * kH * kSKP_VC * kDH * 2);
  bf16* Q_c = (bf16*)take((size_t)kB * kH * kS * kDH * 2);
  bf16* V_c = (bf16*)take((size_t)kB * kH * kSKP_VC * kDH * 2);

  float* km_v = (float*)take(kB * kSKP_VC * 4);
  float* km_q = (float*)take(kB * kSKP_Q * 4);
  float* km_c = (float*)take(kB * kSKP_VC * 4);

  bf16* res_v = (bf16*)take((size_t)kBS * kD * 2);
  bf16* res_q = (bf16*)take((size_t)kBS * kD * 2);
  bf16* res_c = (bf16*)take((size_t)kBS * kD * 2);

  float* out_v = (float*)d_out;
  float* out_q = out_v + (size_t)kBS * kD;
  float* out_c = out_q + (size_t)kBS * kD;

  const int T = 256;
  // 1) means + gates
  k_mean<<<(kB * kD + T - 1) / T, T, 0, stream>>>(v, v_mask, vmean);
  k_mean<<<(kB * kD + T - 1) / T, T, 0, stream>>>(q, q_mask, qmean);
  k_gate<<<(kB * kD + T - 1) / T, T, 0, stream>>>(vmean, v4q_w, v4q_b, g_v4q);
  k_gate<<<(kB * kD + T - 1) / T, T, 0, stream>>>(qmean, q4v_w, q4v_b, g_q4v);

  // 2) bf16 staging
  int nAct = kBS * kD;
  k_relu_bf16<<<(nAct + T - 1) / T, T, 0, stream>>>(v, act_v, nAct);
  k_relu_bf16<<<(nAct + T - 1) / T, T, 0, stream>>>(q, act_q, nAct);
  k_relu_bf16<<<(nAct + T - 1) / T, T, 0, stream>>>(c, act_c, nAct);
  int nLW = kD * kTD, nOW = kD * kD;
  k_transpose_bf16<<<(nLW + T - 1) / T, T, 0, stream>>>(v_lin_w, Wt_v, kD, kTD);
  k_transpose_bf16<<<(nLW + T - 1) / T, T, 0, stream>>>(q_lin_w, Wt_q, kD, kTD);
  k_transpose_bf16<<<(nLW + T - 1) / T, T, 0, stream>>>(c_lin_w, Wt_c, kD, kTD);
  k_transpose_bf16<<<(nOW + T - 1) / T, T, 0, stream>>>(v_out_w, Wt_vo, kD, kD);
  k_transpose_bf16<<<(nOW + T - 1) / T, T, 0, stream>>>(q_out_w, Wt_qo, kD, kD);
  k_transpose_bf16<<<(nOW + T - 1) / T, T, 0, stream>>>(c_out_w, Wt_co, kD, kD);

  // 3) fused QKV projections (WMMA) with gate/mask/scatter epilogue
  dim3 gq(kTD / 256, kBS / 32);
  k_gemm_qkv<<<gq, 256, 0, stream>>>(act_v, Wt_v, v_lin_b, v_mask, g_q4v,
                                     K_v, Q_v, V_v, kSKP_VC, 1);
  k_gemm_qkv<<<gq, 256, 0, stream>>>(act_q, Wt_q, q_lin_b, q_mask, g_v4q,
                                     K_q, Q_q, V_q, kSKP_Q, 0);
  k_gemm_qkv<<<gq, 256, 0, stream>>>(act_c, Wt_c, c_lin_b, c_mask, nullptr,
                                     K_c, Q_c, V_c, kSKP_VC, 1);

  // 4) memory tokens + key masks
  k_memtok<<<(kB * kD + T - 1) / T, T, 0, stream>>>(K_v, V_v, m_v_k, m_v_v, g_q4v, kSKP_VC);
  k_memtok<<<(kB * kD + T - 1) / T, T, 0, stream>>>(K_c, V_c, m_c_k, m_c_v, nullptr, kSKP_VC);
  k_kmask<<<(kB * kSKP_VC + T - 1) / T, T, 0, stream>>>(km_v, v_mask, kSKP_VC, 1);
  k_kmask<<<(kB * kSKP_Q + T - 1) / T, T, 0, stream>>>(km_q, q_mask, kSKP_Q, 0);
  k_kmask<<<(kB * kSKP_VC + T - 1) / T, T, 0, stream>>>(km_c, c_mask, kSKP_VC, 1);

  // 5) flash attention (WMMA), residual fused
  dim3 ga(kS / 64, kH, kB);
  k_attn<<<ga, 128, 0, stream>>>(Q_v, K_v, V_v, km_v, v, res_v, kSKP_VC);
  k_attn<<<ga, 128, 0, stream>>>(Q_q, K_q, V_q, km_q, q, res_q, kSKP_Q);
  k_attn<<<ga, 128, 0, stream>>>(Q_c, K_c, V_c, km_c, c, res_c, kSKP_VC);

  // 6) output projections (WMMA) -> d_out
  dim3 go(kD / 256, kBS / 32);
  k_gemm_out<<<go, 256, 0, stream>>>(res_v, Wt_vo, v_out_b, out_v);
  k_gemm_out<<<go, 256, 0, stream>>>(res_q, Wt_qo, q_out_b, out_q);
  k_gemm_out<<<go, 256, 0, stream>>>(res_c, Wt_co, c_out_b, out_c);
}
